// LanguageWithAttentionSUM_35880156791548
// MI455X (gfx1250) — compile-verified, matrix operand order checked by reading, not run on hardware
//
#include <hip/hip_runtime.h>
#include <hip/hip_bf16.h>

// ---------------------------------------------------------------------------
// LanguageWithAttentionSUM on MI455X (gfx1250, wave32, WMMA).
//
// Roofline: ~17 GFLOP total vs ~0.6 GB HBM traffic dominated by the 328 MB
// scores output -> memory bound. All GEMMs run through
// v_wmma_f32_16x16x32_f16. Weights + embedding table are pre-converted once
// to K-padded f16 (K 300->320, 7->32) so every B-side fragment is a pair of
// guard-free 16B-aligned half loads with zero conversions in the inner
// loops; h / mean / awe are produced directly in f16 by their producers.
// The fc (2048x32x2000) + log_softmax + argmax step is fused: each block
// keeps a 16x2000 f32 logit tile in LDS (128 KB, fits CDNA5's 320 KB/WGP)
// so scores are written to HBM exactly once (non-temporal).
// ---------------------------------------------------------------------------

#define BB   2048
#define VV   2000
#define EMB  300
#define EMBP 320   // padded K for 300-wide operands
#define ENC  128
#define DEC  32
#define TT   20
#define NP   27
#define XW   8275

#define EMB_FULL 288   // 9 full 32-wide K tiles over f32 x rows; tail 288..299

typedef _Float16 half_t;
typedef __attribute__((ext_vector_type(16))) _Float16 v16h;
typedef __attribute__((ext_vector_type(8)))  float    v8f;

__device__ __forceinline__ v8f wmma16(v16h a, v16h b, v8f c) {
  return __builtin_amdgcn_wmma_f32_16x16x32_f16(
      false, a, false, b, (short)0, c, false, false);
}

// 16-bit A/B^T fragment lane pattern per ISA 7.12.2:
//   lane L: row = L%16, khalf = (L/16)*8
//   halves[0..7]  = K = k0 + khalf + j ; halves[8..15] = +16

// f32 source, full tile: unconditional contiguous loads + cvt.
__device__ __forceinline__ v16h frag_full(const float* __restrict__ rowp,
                                          int k0) {
  const float* p = rowp + k0 + (((threadIdx.x & 31) >> 4) << 3);
  v16h r;
#pragma unroll
  for (int j = 0; j < 8; ++j) {
    r[j]     = (_Float16)p[j];
    r[8 + j] = (_Float16)p[16 + j];
  }
  return r;
}

// f32 source, tail tile: index-clamp + value select (v_cndmask), no branches.
__device__ __forceinline__ v16h frag_tail(const float* __restrict__ rowp,
                                          int k0, int kmax) {
  const int kb = ((threadIdx.x & 31) >> 4) << 3;
  v16h r;
#pragma unroll
  for (int j = 0; j < 8; ++j) {
    int ka = k0 + kb + j;
    int kc = ka + 16;
    float fa = rowp[ka < kmax ? ka : 0];
    float fc = rowp[kc < kmax ? kc : 0];
    r[j]     = (_Float16)(ka < kmax ? fa : 0.0f);
    r[8 + j] = (_Float16)(kc < kmax ? fc : 0.0f);
  }
  return r;
}

// f16 source (pre-padded): pure contiguous half loads, no cvt, no guards.
__device__ __forceinline__ v16h frag_h(const half_t* __restrict__ rowp,
                                       int k0) {
  const half_t* p = rowp + k0 + (((threadIdx.x & 31) >> 4) << 3);
  v16h r;
#pragma unroll
  for (int j = 0; j < 8; ++j) {
    r[j]     = p[j];
    r[8 + j] = p[16 + j];
  }
  return r;
}

// D tile: lane L -> col n0 + L%16, rows m0 + (L/16)*8 + v
// mode: 0 = identity, 1 = relu, 2 = sigmoid
__device__ __forceinline__ void store_tile(float* __restrict__ dst, int ld,
                                           int m0, int n0, v8f acc,
                                           const float* __restrict__ bias,
                                           int mode) {
  const int lane = threadIdx.x & 31;
  const int n = n0 + (lane & 15);
  const int mb = (lane >> 4) << 3;
  const float bv = bias ? bias[n] : 0.0f;
#pragma unroll
  for (int v = 0; v < 8; ++v) {
    float val = acc[v] + bv;
    if (mode == 1) val = fmaxf(val, 0.0f);
    else if (mode == 2) val = 1.0f / (1.0f + __expf(-val));
    dst[(size_t)(m0 + mb + v) * ld + n] = val;
  }
}

// -------------------- one-time f32 -> padded-f16 conversion ----------------

__global__ void cvt_pad_kernel(half_t* __restrict__ dst,
                               const float* __restrict__ src,
                               int srcStride, int ksrc, int kpad, int total) {
  int idx = blockIdx.x * blockDim.x + threadIdx.x;
  if (idx >= total) return;
  int r = idx / kpad, k = idx % kpad;
  dst[idx] = (half_t)(k < ksrc ? src[(size_t)r * srcStride + k] : 0.0f);
}

// --------------------------- encoder kernels -------------------------------

// grid rows: enc[b,p,:] = relu(grid_emb@We^T+be) + relu(grid_oh@Wo^T+bo)
__global__ void grid_enc_kernel(const float* __restrict__ x,
                                const half_t* __restrict__ We16,
                                const float* __restrict__ b_embed,
                                const half_t* __restrict__ Wo16,
                                const float* __restrict__ b_onehot,
                                float* __restrict__ enc) {
  const int r0 = blockIdx.x * 16;           // row tile over B*25 rows
  const int n0 = blockIdx.y * 16;
  const int lane = threadIdx.x & 31;
  const int mr = r0 + (lane & 15);
  const int b = mr / 25, p = mr % 25;
  const float* rowEmb = x + (size_t)b * XW + (size_t)p * EMB;
  const float* rowOh  = x + (size_t)b * XW + 7500 + (size_t)p * 7;
  const half_t* w1 = We16 + (size_t)(n0 + (lane & 15)) * EMBP;
  const half_t* w2 = Wo16 + (size_t)(n0 + (lane & 15)) * 32;

  v8f acc1 = {}, acc2 = {};
  for (int k0 = 0; k0 < EMB_FULL; k0 += 32) {
    __builtin_prefetch(rowEmb + k0 + 64, 0, 3);   // global_prefetch_b8
    acc1 = wmma16(frag_full(rowEmb, k0), frag_h(w1, k0), acc1);
  }
  acc1 = wmma16(frag_tail(rowEmb, EMB_FULL, EMB), frag_h(w1, EMB_FULL), acc1);
  acc2 = wmma16(frag_tail(rowOh, 0, 7), frag_h(w2, 0), acc2);

  const int n = n0 + (lane & 15);
  const int mb = (lane >> 4) << 3;
  const float b1 = b_embed[n], b2 = b_onehot[n];
#pragma unroll
  for (int v = 0; v < 8; ++v) {
    int r = r0 + mb + v;
    int bb2 = r / 25, pp = r % 25;
    float val = fmaxf(acc1[v] + b1, 0.0f) + fmaxf(acc2[v] + b2, 0.0f);
    enc[((size_t)bb2 * NP + pp) * ENC + n] = val;
  }
}

// inv (p=25) / goal (p=26) rows
__global__ void invgoal_kernel(const float* __restrict__ x,
                               const half_t* __restrict__ Winv16,
                               const float* __restrict__ b_inv,
                               const half_t* __restrict__ Wgoal16,
                               const float* __restrict__ b_goal,
                               float* __restrict__ enc) {
  const int m0 = blockIdx.x * 16;
  const int n0 = blockIdx.y * 16;
  const int t = blockIdx.z;                 // 0=inv, 1=goal
  const int lane = threadIdx.x & 31;
  const int b = m0 + (lane & 15);
  const half_t* W = t ? Wgoal16 : Winv16;
  const float* bias = t ? b_goal : b_inv;
  const float* rowp = x + (size_t)b * XW + (t ? 7675 : 7975);
  const half_t* wrow = W + (size_t)(n0 + (lane & 15)) * EMBP;

  v8f acc = {};
  for (int k0 = 0; k0 < EMB_FULL; k0 += 32)
    acc = wmma16(frag_full(rowp, k0), frag_h(wrow, k0), acc);
  acc = wmma16(frag_tail(rowp, EMB_FULL, EMB), frag_h(wrow, EMB_FULL), acc);

  const int n = n0 + (lane & 15);
  const int mb = (lane >> 4) << 3;
  const float bv = bias[n];
#pragma unroll
  for (int v = 0; v < 8; ++v) {
    int bb2 = m0 + mb + v;
    enc[((size_t)bb2 * NP + (25 + t)) * ENC + n] = fmaxf(acc[v] + bv, 0.0f);
  }
}

__global__ void mean_kernel(const float* __restrict__ enc,
                            half_t* __restrict__ mean16) {
  int idx = blockIdx.x * blockDim.x + threadIdx.x;  // B*ENC
  if (idx >= BB * ENC) return;
  int b = idx / ENC, e = idx % ENC;
  float s = 0.0f;
#pragma unroll
  for (int p = 0; p < NP; ++p) s += enc[((size_t)b * NP + p) * ENC + e];
  mean16[idx] = (half_t)(s * (1.0f / (float)NP));
}

// h0 = mean@W_h0^T + b_h0 (-> f16 h) ; c0 = mean@W_c0^T + b_c0 (-> f32 c)
__global__ void h0c0_kernel(const half_t* __restrict__ mean16,
                            const half_t* __restrict__ Wh016,
                            const float* __restrict__ b_h0,
                            const half_t* __restrict__ Wc016,
                            const float* __restrict__ b_c0,
                            half_t* __restrict__ h16,
                            float* __restrict__ cbuf) {
  const int m0 = blockIdx.x * 16;
  const int n0 = blockIdx.y * 16;
  const int z = blockIdx.z;
  const int lane = threadIdx.x & 31;
  const half_t* W = z ? Wc016 : Wh016;
  const float* bias = z ? b_c0 : b_h0;
  const half_t* rowp = mean16 + (size_t)(m0 + (lane & 15)) * ENC;
  const half_t* wrow = W + (size_t)(n0 + (lane & 15)) * ENC;
  v8f acc = {};
#pragma unroll
  for (int k0 = 0; k0 < ENC; k0 += 32)
    acc = wmma16(frag_h(rowp, k0), frag_h(wrow, k0), acc);

  const int n = n0 + (lane & 15);
  const int mb = (lane >> 4) << 3;
  const float bv = bias[n];
#pragma unroll
  for (int v = 0; v < 8; ++v) {
    float val = acc[v] + bv;
    if (z) cbuf[(size_t)(m0 + mb + v) * DEC + n] = val;
    else   h16[(size_t)(m0 + mb + v) * DEC + n] = (half_t)val;
  }
}

// att1 = enc @ W_ea^T + b_ea   (B*27 x 128, K=128)
__global__ void att1_kernel(const float* __restrict__ enc,
                            const half_t* __restrict__ Wea16,
                            const float* __restrict__ b_ea,
                            float* __restrict__ att1) {
  const int m0 = blockIdx.x * 16;
  const int n0 = blockIdx.y * 16;
  const int lane = threadIdx.x & 31;
  const float* rowp = enc + (size_t)(m0 + (lane & 15)) * ENC;
  const half_t* wrow = Wea16 + (size_t)(n0 + (lane & 15)) * ENC;
  v8f acc = {};
#pragma unroll
  for (int k0 = 0; k0 < ENC; k0 += 32) {
    __builtin_prefetch(rowp + k0 + 64, 0, 3);
    acc = wmma16(frag_full(rowp, k0), frag_h(wrow, k0), acc);
  }
  store_tile(att1, ENC, m0, n0, acc, b_ea, 0);
}

// --------------------------- decoder kernels -------------------------------

__global__ void init_prev_kernel(int* __restrict__ prev,
                                 const int* __restrict__ start_idx) {
  int i = blockIdx.x * blockDim.x + threadIdx.x;
  if (i < BB) prev[i] = start_idx[0];
}

// att2 = h@W_da^T + b_da ; gate = sigmoid(h@W_beta^T + b_beta)   (K=32)
__global__ void att2gate_kernel(const half_t* __restrict__ h16,
                                const half_t* __restrict__ Wda16,
                                const float* __restrict__ b_da,
                                const half_t* __restrict__ Wbeta16,
                                const float* __restrict__ b_beta,
                                float* __restrict__ att2,
                                float* __restrict__ gateb) {
  const int m0 = blockIdx.x * 16;
  const int n0 = blockIdx.y * 16;
  const int z = blockIdx.z;
  const int lane = threadIdx.x & 31;
  const half_t* W = z ? Wbeta16 : Wda16;
  const float* bias = z ? b_beta : b_da;
  float* out = z ? gateb : att2;
  const half_t* rowp = h16 + (size_t)(m0 + (lane & 15)) * DEC;
  const half_t* wrow = W + (size_t)(n0 + (lane & 15)) * DEC;
  v8f acc = {};
  acc = wmma16(frag_h(rowp, 0), frag_h(wrow, 0), acc);
  store_tile(out, ENC, m0, n0, acc, bias, z ? 2 : 0);
}

// attention scores + softmax + weighted enc sum; one wave per batch item
__global__ void attention_kernel(const float* __restrict__ att1,
                                 const float* __restrict__ att2,
                                 const float* __restrict__ enc,
                                 const float* __restrict__ gateb,
                                 const float* __restrict__ W_fa,
                                 const float* __restrict__ b_fa,
                                 half_t* __restrict__ awe16) {
  const int b = blockIdx.x;
  const int lane = threadIdx.x;             // 32 lanes, 4 channels each
  const float4 a2 = ((const float4*)(att2 + (size_t)b * ENC))[lane];
  const float4 wf = ((const float4*)W_fa)[lane];
  float att[NP];
#pragma unroll
  for (int p = 0; p < NP; ++p) {
    const float4 a1 = ((const float4*)(att1 + ((size_t)b * NP + p) * ENC))[lane];
    float part = fmaxf(a1.x + a2.x, 0.0f) * wf.x
               + fmaxf(a1.y + a2.y, 0.0f) * wf.y
               + fmaxf(a1.z + a2.z, 0.0f) * wf.z
               + fmaxf(a1.w + a2.w, 0.0f) * wf.w;
#pragma unroll
    for (int off = 16; off >= 1; off >>= 1) part += __shfl_xor(part, off, 32);
    att[p] = part + b_fa[0];
  }
  float mx = att[0];
#pragma unroll
  for (int p = 1; p < NP; ++p) mx = fmaxf(mx, att[p]);
  float s = 0.0f;
#pragma unroll
  for (int p = 0; p < NP; ++p) { att[p] = __expf(att[p] - mx); s += att[p]; }
  const float invs = 1.0f / s;
  float4 acc = {0.0f, 0.0f, 0.0f, 0.0f};
#pragma unroll
  for (int p = 0; p < NP; ++p) {
    const float4 e = ((const float4*)(enc + ((size_t)b * NP + p) * ENC))[lane];
    const float al = att[p] * invs;
    acc.x += al * e.x; acc.y += al * e.y; acc.z += al * e.z; acc.w += al * e.w;
  }
  const float4 g = ((const float4*)(gateb + (size_t)b * ENC))[lane];
  half_t* o = awe16 + (size_t)b * ENC + 4 * lane;
  o[0] = (half_t)(acc.x * g.x); o[1] = (half_t)(acc.y * g.y);
  o[2] = (half_t)(acc.z * g.z); o[3] = (half_t)(acc.w * g.w);
}

// gates = E[prev]@Wih[:, :300]^T + awe@Wih[:,300:]^T + h@Whh^T + b_ih + b_hh
__global__ void gates_kernel(const half_t* __restrict__ E16,
                             const int* __restrict__ prev,
                             const half_t* __restrict__ awe16,
                             const half_t* __restrict__ h16,
                             const half_t* __restrict__ WihA16,
                             const float* __restrict__ b_ih,
                             const half_t* __restrict__ WihB16,
                             const half_t* __restrict__ Whh16,
                             const float* __restrict__ b_hh,
                             float* __restrict__ gatesb) {
  const int m0 = blockIdx.x * 16;
  const int n0 = blockIdx.y * 16;
  const int lane = threadIdx.x & 31;
  const int b = m0 + (lane & 15);
  const int nrow = n0 + (lane & 15);
  const half_t* rowEmb = E16 + (size_t)prev[b] * EMBP;   // padded, guard-free
  const half_t* rowAwe = awe16 + (size_t)b * ENC;
  const half_t* rowH   = h16 + (size_t)b * DEC;
  const half_t* wA = WihA16 + (size_t)nrow * EMBP;
  const half_t* wB = WihB16 + (size_t)nrow * ENC;
  const half_t* wH = Whh16 + (size_t)nrow * DEC;

  v8f acc = {};
#pragma unroll
  for (int k0 = 0; k0 < EMBP; k0 += 32)
    acc = wmma16(frag_h(rowEmb, k0), frag_h(wA, k0), acc);
#pragma unroll
  for (int k0 = 0; k0 < ENC; k0 += 32)
    acc = wmma16(frag_h(rowAwe, k0), frag_h(wB, k0), acc);
  acc = wmma16(frag_h(rowH, 0), frag_h(wH, 0), acc);

  const int n = n0 + (lane & 15);
  const int mb = (lane >> 4) << 3;
  const float bv = b_ih[n] + b_hh[n];
#pragma unroll
  for (int v = 0; v < 8; ++v)
    gatesb[(size_t)(m0 + mb + v) * ENC + n] = acc[v] + bv;
}

__global__ void lstm_kernel(const float* __restrict__ gatesb,
                            half_t* __restrict__ h16,
                            float* __restrict__ cbuf) {
  int idx = blockIdx.x * blockDim.x + threadIdx.x;   // B*DEC
  if (idx >= BB * DEC) return;
  int b = idx / DEC, j = idx % DEC;
  const float* g = gatesb + (size_t)b * ENC;
  float ig = g[j], fg = g[DEC + j], gg = g[2 * DEC + j], og = g[3 * DEC + j];
  float si = 1.0f / (1.0f + __expf(-ig));
  float sf = 1.0f / (1.0f + __expf(-fg));
  float so = 1.0f / (1.0f + __expf(-og));
  float cn = sf * cbuf[idx] + si * tanhf(gg);
  cbuf[idx] = cn;
  h16[idx] = (half_t)(so * tanhf(cn));
}

// Fused fc GEMM (B x 32 x 2000) + log_softmax + first-occurrence argmax.
// 8 waves / block, 16 batch rows / block; 16x2000 f32 logits in LDS (128 KB).
__global__ void fc_softmax_kernel(const half_t* __restrict__ h16,
                                  const half_t* __restrict__ Wfc16,
                                  const float* __restrict__ b_fc,
                                  float* __restrict__ scores,
                                  float* __restrict__ seqs,
                                  int* __restrict__ prev, int t) {
  extern __shared__ float logits[];                // 16 * VV floats
  const int b0 = blockIdx.x * 16;
  const int wave = threadIdx.x >> 5;
  const int lane = threadIdx.x & 31;

  const half_t* arow = h16 + (size_t)(b0 + (lane & 15)) * DEC;
  v16h a = frag_h(arow, 0);                        // reused across 16 tiles
  for (int tt = 0; tt < 16; ++tt) {
    int tile = wave * 16 + tt;                     // 125 valid col tiles
    if (tile >= 125) break;                        // wave-uniform branch
    int n0 = tile * 16;
    const half_t* wrow = Wfc16 + (size_t)(n0 + (lane & 15)) * DEC;
    v16h bf = frag_h(wrow, 0);
    v8f acc = {};
    acc = wmma16(a, bf, acc);
    const int n = n0 + (lane & 15);
    const int mb = (lane >> 4) << 3;
    const float bv = b_fc[n];
#pragma unroll
    for (int v = 0; v < 8; ++v)
      logits[(mb + v) * VV + n] = acc[v] + bv;
  }
  __syncthreads();

  // each wave normalizes 2 rows
#pragma unroll
  for (int rr = 0; rr < 2; ++rr) {
    const int row = wave * 2 + rr;
    const float* lrow = logits + (size_t)row * VV;
    float bestv = -1e30f; int besti = 0;
    for (int c = lane; c < VV; c += 32) {
      float v = lrow[c];
      if (v > bestv) { bestv = v; besti = c; }     // lane-local first max
    }
#pragma unroll
    for (int off = 16; off >= 1; off >>= 1) {
      float ov = __shfl_xor(bestv, off, 32);
      int oi = __shfl_xor(besti, off, 32);
      if (ov > bestv || (ov == bestv && oi < besti)) { bestv = ov; besti = oi; }
    }
    float s = 0.0f;
    for (int c = lane; c < VV; c += 32) s += __expf(lrow[c] - bestv);
#pragma unroll
    for (int off = 16; off >= 1; off >>= 1) s += __shfl_xor(s, off, 32);
    const float logZ = bestv + __logf(s);
    const size_t obase = ((size_t)t * BB + (b0 + row)) * VV;
    for (int c = lane; c < VV; c += 32)
      __builtin_nontemporal_store(lrow[c] - logZ, &scores[obase + c]);
    if (lane == 0) {
      seqs[(size_t)t * BB + (b0 + row)] = (float)besti;
      prev[b0 + row] = besti;
    }
  }
}

// --------------------------------- host ------------------------------------

extern "C" void kernel_launch(void* const* d_in, const int* in_sizes, int n_in,
                              void* d_out, int out_size, void* d_ws, size_t ws_size,
                              hipStream_t stream) {
  const float* x        = (const float*)d_in[0];
  const float* E        = (const float*)d_in[1];
  const float* W_embed  = (const float*)d_in[2];
  const float* b_embed  = (const float*)d_in[3];
  const float* W_onehot = (const float*)d_in[4];
  const float* b_onehot = (const float*)d_in[5];
  const float* W_inv    = (const float*)d_in[6];
  const float* b_inv    = (const float*)d_in[7];
  const float* W_goal   = (const float*)d_in[8];
  const float* b_goal   = (const float*)d_in[9];
  const float* W_ea     = (const float*)d_in[10];
  const float* b_ea     = (const float*)d_in[11];
  const float* W_da     = (const float*)d_in[12];
  const float* b_da     = (const float*)d_in[13];
  const float* W_fa     = (const float*)d_in[14];
  const float* b_fa     = (const float*)d_in[15];
  const float* W_ih     = (const float*)d_in[16];
  const float* b_ih     = (const float*)d_in[17];
  const float* W_hh     = (const float*)d_in[18];
  const float* b_hh     = (const float*)d_in[19];
  const float* W_h0     = (const float*)d_in[20];
  const float* b_h0     = (const float*)d_in[21];
  const float* W_c0     = (const float*)d_in[22];
  const float* b_c0     = (const float*)d_in[23];
  const float* W_beta   = (const float*)d_in[24];
  const float* b_beta   = (const float*)d_in[25];
  const float* W_fc     = (const float*)d_in[26];
  const float* b_fc     = (const float*)d_in[27];
  const int*   start    = (const int*)d_in[28];

  float* scores = (float*)d_out;                         // (T, B, V)
  float* seqs   = scores + (size_t)TT * BB * VV;         // (T, B) as float

  // 256B-aligned bump allocator over d_ws (~60 MB total)
  char* wsb = (char*)d_ws;
  size_t off = 0;
  auto alloc = [&](size_t bytes) -> void* {
    void* p = wsb + off;
    off = (off + bytes + 255) & ~(size_t)255;
    return p;
  };
  float* enc    = (float*)alloc((size_t)BB * NP * ENC * 4);
  float* att1   = (float*)alloc((size_t)BB * NP * ENC * 4);
  float* cbuf   = (float*)alloc((size_t)BB * DEC * 4);
  float* att2   = (float*)alloc((size_t)BB * ENC * 4);
  float* gateb  = (float*)alloc((size_t)BB * ENC * 4);
  float* gatesb = (float*)alloc((size_t)BB * ENC * 4);
  int*   prev   = (int*)alloc((size_t)BB * 4);
  half_t* mean16 = (half_t*)alloc((size_t)BB * ENC * 2);
  half_t* h16    = (half_t*)alloc((size_t)BB * DEC * 2);
  half_t* awe16  = (half_t*)alloc((size_t)BB * ENC * 2);
  half_t* E16    = (half_t*)alloc((size_t)2001 * EMBP * 2);
  half_t* We16   = (half_t*)alloc((size_t)ENC * EMBP * 2);
  half_t* Wo16   = (half_t*)alloc((size_t)ENC * 32 * 2);
  half_t* Winv16 = (half_t*)alloc((size_t)ENC * EMBP * 2);
  half_t* Wgoal16= (half_t*)alloc((size_t)ENC * EMBP * 2);
  half_t* Wea16  = (half_t*)alloc((size_t)ENC * ENC * 2);
  half_t* Wda16  = (half_t*)alloc((size_t)ENC * DEC * 2);
  half_t* Wbeta16= (half_t*)alloc((size_t)ENC * DEC * 2);
  half_t* Wh016  = (half_t*)alloc((size_t)DEC * ENC * 2);
  half_t* Wc016  = (half_t*)alloc((size_t)DEC * ENC * 2);
  half_t* WihA16 = (half_t*)alloc((size_t)ENC * EMBP * 2);
  half_t* WihB16 = (half_t*)alloc((size_t)ENC * ENC * 2);
  half_t* Whh16  = (half_t*)alloc((size_t)ENC * DEC * 2);
  half_t* Wfc16  = (half_t*)alloc((size_t)VV * DEC * 2);

  // ---- one-time weight/table conversions to padded f16 ----
  auto cvt = [&](half_t* dst, const float* src, int stride, int ksrc,
                 int kpad, int rows) {
    int total = rows * kpad;
    cvt_pad_kernel<<<(total + 255) / 256, 256, 0, stream>>>(
        dst, src, stride, ksrc, kpad, total);
  };
  cvt(E16,    E,        EMB, EMB, EMBP, 2001);
  cvt(We16,   W_embed,  EMB, EMB, EMBP, ENC);
  cvt(Wo16,   W_onehot, 7,   7,   32,   ENC);
  cvt(Winv16, W_inv,    EMB, EMB, EMBP, ENC);
  cvt(Wgoal16,W_goal,   EMB, EMB, EMBP, ENC);
  cvt(Wea16,  W_ea,     ENC, ENC, ENC,  ENC);
  cvt(Wda16,  W_da,     DEC, DEC, DEC,  ENC);
  cvt(Wbeta16,W_beta,   DEC, DEC, DEC,  ENC);
  cvt(Wh016,  W_h0,     ENC, ENC, ENC,  DEC);
  cvt(Wc016,  W_c0,     ENC, ENC, ENC,  DEC);
  cvt(WihA16, W_ih,          EMB + ENC, EMB, EMBP, ENC);
  cvt(WihB16, W_ih + EMB,    EMB + ENC, ENC, ENC,  ENC);
  cvt(Whh16,  W_hh,     DEC, DEC, DEC,  ENC);
  cvt(Wfc16,  W_fc,     DEC, DEC, DEC,  VV);

  // ---- encoder ----
  init_prev_kernel<<<(BB + 255) / 256, 256, 0, stream>>>(prev, start);
  grid_enc_kernel<<<dim3(BB * 25 / 16, ENC / 16), 32, 0, stream>>>(
      x, We16, b_embed, Wo16, b_onehot, enc);
  invgoal_kernel<<<dim3(BB / 16, ENC / 16, 2), 32, 0, stream>>>(
      x, Winv16, b_inv, Wgoal16, b_goal, enc);
  mean_kernel<<<(BB * ENC + 255) / 256, 256, 0, stream>>>(enc, mean16);
  h0c0_kernel<<<dim3(BB / 16, DEC / 16, 2), 32, 0, stream>>>(
      mean16, Wh016, b_h0, Wc016, b_c0, h16, cbuf);
  att1_kernel<<<dim3(BB * NP / 16, ENC / 16), 32, 0, stream>>>(
      enc, Wea16, b_ea, att1);

  // ---- decoder: host-sequential T loop (argmax feedback dependency) ----
  const size_t fc_lds = (size_t)16 * VV * sizeof(float);  // 128 KB of 320 KB/WGP
  for (int t = 0; t < TT; ++t) {
    att2gate_kernel<<<dim3(BB / 16, ENC / 16, 2), 32, 0, stream>>>(
        h16, Wda16, b_da, Wbeta16, b_beta, att2, gateb);
    attention_kernel<<<BB, 32, 0, stream>>>(
        att1, att2, enc, gateb, W_fa, b_fa, awe16);
    gates_kernel<<<dim3(BB / 16, ENC / 16), 32, 0, stream>>>(
        E16, prev, awe16, h16, WihA16, b_ih, WihB16, Whh16, b_hh, gatesb);
    lstm_kernel<<<(BB * DEC + 255) / 256, 256, 0, stream>>>(gatesb, h16, cbuf);
    fc_softmax_kernel<<<BB / 16, 256, fc_lds, stream>>>(
        h16, Wfc16, b_fc, scores, seqs, prev, t);
  }

  (void)in_sizes; (void)n_in; (void)out_size; (void)ws_size;
}
